// CausalSelfAttention_58729382805737
// MI455X (gfx1250) — matrix operand order repaired and checked by reading, not compiled, not run on hardware
//
#include <hip/hip_runtime.h>
#include <hip/hip_bf16.h>

typedef __attribute__((ext_vector_type(8)))  _Float16 v8h;
typedef __attribute__((ext_vector_type(16))) _Float16 v16h;
typedef __attribute__((ext_vector_type(8)))  float    v8f;

#define T_SEQ 2048
#define C_DIM 1024
#define N_HEAD 16
#define D_HEAD 64
#define M_ROWS 8192   // B*T

// flash-attention dynamic-LDS layout (bytes); stride 80 halves = 160B keeps
// every fragment row 16B-aligned and spreads banks.  K/V double-buffered.
#define KV_TILE_BYTES 10240            // 64 rows x 80 halves x 2B
#define KOFF(buf) ((buf) * 2 * KV_TILE_BYTES)
#define VOFF(buf) ((buf) * 2 * KV_TILE_BYTES + KV_TILE_BYTES)
#define POFF 40960
#define SMEM_BYTES 61440               // 2*(K+V) + 8-wave P staging

// ---------------------------------------------------------------------------
// Fragment loaders matching CDNA5 wave32 WMMA VGPR layouts (CDNA5 ISA)
// A 16x32 f16 (MxK): lanes 0-15 hold row M=lane, K=[kh..kh+7],[kh+16..kh+23],
//                    kh = (lane>=16)*8.  Row-major source, ld in halves.
// B 32x16 f16 (KxN): lane holds column N=lane&15, K=[ko..ko+7],[ko+16..ko+23],
//                    ko=(lane>=16)*8.  Source must be K-contiguous per column.
// C/D 16x16 f32:     VGPR r holds row M=r (lanes0-15) / M=8+r (lanes16-31),
//                    N = lane&15.
// ---------------------------------------------------------------------------
__device__ __forceinline__ v16h make_v16(v8h lo, v8h hi) {
  v16h r;
#pragma unroll
  for (int i = 0; i < 8; ++i) { r[i] = lo[i]; r[i + 8] = hi[i]; }
  return r;
}

__device__ __forceinline__ v16h load_frag_a(const _Float16* p, int ld) {
  int lane = threadIdx.x & 31;
  int m  = lane & 15;
  int kh = (lane >> 4) << 3;               // 0 or 8
  const _Float16* q = p + (size_t)m * ld + kh;
  v8h lo = *(const v8h*)(q);
  v8h hi = *(const v8h*)(q + 16);
  return make_v16(lo, hi);
}

__device__ __forceinline__ v16h load_frag_b(const _Float16* p, int ld) {
  int lane = threadIdx.x & 31;
  int n  = lane & 15;
  int ko = (lane >> 4) << 3;               // 0 or 8
  const _Float16* q = p + (size_t)n * ld + ko;
  v8h lo = *(const v8h*)(q);
  v8h hi = *(const v8h*)(q + 16);
  return make_v16(lo, hi);
}

__device__ __forceinline__ v8f wmma_step(v16h a, v16h b, v8f c) {
  return __builtin_amdgcn_wmma_f32_16x16x32_f16(
      /*neg_a=*/false, a, /*neg_b=*/false, b,
      /*c_mod=*/(short)0, c, /*reuse_a=*/false, /*reuse_b=*/false);
}

// Async global->LDS copy of one 16B chunk (per lane), ASYNCcnt-tracked.
__device__ __forceinline__ void async_copy_b128(uint32_t lds_off, const void* gptr) {
  uint64_t ga = (uint64_t)gptr;
  asm volatile("global_load_async_to_lds_b128 %0, %1, off"
               :: "v"(lds_off), "v"(ga) : "memory");
}
// Async loads complete in order: waiting to <=N leaves the newest N in flight.
__device__ __forceinline__ void wait_async_le4() {
  asm volatile("s_wait_asynccnt 0x4" ::: "memory");
}
__device__ __forceinline__ void wait_async_le0() {
  asm volatile("s_wait_asynccnt 0x0" ::: "memory");
}

// ---------------------------------------------------------------------------
// fp32 -> f16 downcast (x and the four weight matrices)
// ---------------------------------------------------------------------------
__global__ void cvt_f16_kernel(const float* __restrict__ in,
                               _Float16* __restrict__ out, long long n) {
  long long i = (long long)blockIdx.x * blockDim.x + threadIdx.x;
  long long stride = (long long)gridDim.x * blockDim.x;
  for (; i < n; i += stride) out[i] = (_Float16)in[i];
}

// ---------------------------------------------------------------------------
// C[M,N] = A[M,K](f16) * W[N,K](f16)^T + bias ; f16 or f32 output.
// 256 threads = 8 waves (2x4); WG tile 128x128; wave tile 64x32
// (4x2 accumulators -> 8 WMMAs per 12 b128 loads per k-step).
// ---------------------------------------------------------------------------
__global__ __launch_bounds__(256)
void gemm_bias_kernel(const _Float16* __restrict__ A,
                      const _Float16* __restrict__ W,
                      const float* __restrict__ bias,
                      _Float16* __restrict__ outh,
                      float* __restrict__ outf,
                      int M, int N, int K, int f32out) {
  int wave = threadIdx.x >> 5;
  int lane = threadIdx.x & 31;
  int wm = wave & 1, wn = wave >> 1;
  int mbase = blockIdx.y * 128 + wm * 64;
  int nbase = blockIdx.x * 128 + wn * 32;

  const _Float16* Ap = A + (size_t)mbase * K;
  const _Float16* Wp = W + (size_t)nbase * K;

  v8f acc[4][2] = {};
  for (int k0 = 0; k0 < K; k0 += 32) {
    __builtin_prefetch(Ap + k0 + 512, 0, 1);          // global_prefetch_b8
    __builtin_prefetch(Wp + k0 + 512, 0, 1);
    v16h a[4], b[2];
#pragma unroll
    for (int mi = 0; mi < 4; ++mi)
      a[mi] = load_frag_a(Ap + (size_t)(mi * 16) * K + k0, K);
#pragma unroll
    for (int ni = 0; ni < 2; ++ni)
      b[ni] = load_frag_b(Wp + (size_t)(ni * 16) * K + k0, K);
#pragma unroll
    for (int mi = 0; mi < 4; ++mi)
#pragma unroll
      for (int ni = 0; ni < 2; ++ni)
        acc[mi][ni] = wmma_step(a[mi], b[ni], acc[mi][ni]);
  }

  int lc = lane & 15, ro = (lane >> 4) * 8;
#pragma unroll
  for (int mi = 0; mi < 4; ++mi)
#pragma unroll
    for (int ni = 0; ni < 2; ++ni) {
      int col = nbase + ni * 16 + lc;
      float bv = bias[col];
#pragma unroll
      for (int r = 0; r < 8; ++r) {
        int row = mbase + mi * 16 + r + ro;
        float val = acc[mi][ni][r] + bv;
        if (f32out) outf[(size_t)row * N + col] = val;
        else        outh[(size_t)row * N + col] = (_Float16)val;
      }
    }
}

// ---------------------------------------------------------------------------
// RoPE + head split.  qf/kf/vf are [B,T,H*64] f16 (proj outputs).
// Q,Kr -> [B,H,T,64] interleaved rotation;  Vt -> [B,H,64,T] transposed so
// PV B-fragments are key-contiguous.  One thread per (b,h,t,j), j=0..31.
// ---------------------------------------------------------------------------
__global__ void rope_split_kernel(const _Float16* __restrict__ qf,
                                  const _Float16* __restrict__ kf,
                                  const _Float16* __restrict__ vf,
                                  _Float16* __restrict__ Q,
                                  _Float16* __restrict__ Kr,
                                  _Float16* __restrict__ Vt) {
  int idx = blockIdx.x * 256 + threadIdx.x;      // exact multiple of total
  int j  = idx & 31;
  int t  = (idx >> 5) & (T_SEQ - 1);
  int bh = idx >> 16;                            // 32*2048 = 2^16
  int h  = bh & (N_HEAD - 1);
  int b  = bh >> 4;

  size_t src = ((size_t)(b * T_SEQ + t)) * C_DIM + h * D_HEAD + 2 * j;
  size_t dq  = ((size_t)bh * T_SEQ + t) * D_HEAD + 2 * j;

  float theta = __powf(10000.0f, -(float)(2 * j) / (float)D_HEAD);
  float ang = (float)t * theta;
  float s, c;
  sincosf(ang, &s, &c);

  float q0 = (float)qf[src], q1 = (float)qf[src + 1];
  Q[dq]     = (_Float16)(q0 * c - q1 * s);
  Q[dq + 1] = (_Float16)(q0 * s + q1 * c);

  float k0 = (float)kf[src], k1 = (float)kf[src + 1];
  Kr[dq]     = (_Float16)(k0 * c - k1 * s);
  Kr[dq + 1] = (_Float16)(k0 * s + k1 * c);

  size_t dv = ((size_t)bh * D_HEAD + 2 * j) * T_SEQ + t;
  Vt[dv]         = vf[src];
  Vt[dv + T_SEQ] = vf[src + 1];
}

// ---------------------------------------------------------------------------
// Causal flash attention, cooperative + double-buffered.
// Q,K: [B,H,T,64] f16; Vt: [B,H,64,T] f16.  8 waves share a 128-row Q block
// and iterate the same 64-key blocks.  K/V tiles ping-pong through LDS via
// global_load_async_to_lds_b128: block i+1's copies are issued before block i
// is computed, and s_wait_asynccnt 0x4 (in-order completion) retires only the
// older block's transfers.  Row stats via shfl_xor within 16-lane halves
// (wave32); P re-laid-out C->A through wave-private padded LDS.
// ---------------------------------------------------------------------------
__global__ __launch_bounds__(256)
void flash_attn_kernel(const _Float16* __restrict__ Q,
                       const _Float16* __restrict__ Kk,
                       const _Float16* __restrict__ Vt,
                       _Float16* __restrict__ Y) {
  extern __shared__ __align__(16) char smem[];
  _Float16* pw = (_Float16*)(smem + POFF);               // [8][16*80]

  int wave = threadIdx.x >> 5;
  int lane = threadIdx.x & 31;
  int tid  = threadIdx.x;
  int bh = blockIdx.y;
  int b = bh >> 4, h = bh & (N_HEAD - 1);
  int qbase = blockIdx.x * 128 + wave * 16;
  int qtop  = blockIdx.x * 128 + 127;
  int lc = lane & 15, ro = (lane >> 4) * 8;

  const _Float16* qp = Q + ((size_t)bh * T_SEQ + qbase) * D_HEAD;
  v16h aq0 = load_frag_a(qp,      D_HEAD);   // d = 0..31
  v16h aq1 = load_frag_a(qp + 32, D_HEAD);   // d = 32..63

  v8f o[4] = {};
  float mrow[8], lrow[8];
#pragma unroll
  for (int r = 0; r < 8; ++r) { mrow[r] = -3.0e38f; lrow[r] = 0.0f; }

  _Float16* pwv = pw + wave * 16 * 80;

  // per-thread staging map: 256 threads x (2 rows) x 16B cover a 64x64 f16 tile
  int srow = tid >> 3;                 // 0..31
  int sch  = (tid & 7) * 8;            // half offset within row (8h = 16B)
  const _Float16* kgbase = Kk + (size_t)bh * T_SEQ * D_HEAD;
  const _Float16* vgbase = Vt + (size_t)bh * D_HEAD * T_SEQ;

  // stage key block starting at kb into buffer buf (4 async b128 per thread)
  auto stage = [&](int buf, int kb) {
#pragma unroll
    for (int hlf = 0; hlf < 2; ++hlf) {
      int r = srow + hlf * 32;
      async_copy_b128((uint32_t)(KOFF(buf) + (r * 80 + sch) * 2),
                      kgbase + (size_t)(kb + r) * D_HEAD + sch);
      async_copy_b128((uint32_t)(VOFF(buf) + (r * 80 + sch) * 2),
                      vgbase + (size_t)r * T_SEQ + kb + sch);
    }
  };

  int nblk = (qtop >> 6) + 1;          // 64-key blocks this WG visits
  stage(0, 0);

  for (int ib = 0; ib < nblk; ++ib) {
    int kb  = ib << 6;
    int cur = ib & 1;
    bool more = (ib + 1) < nblk;
    if (more) stage(cur ^ 1, kb + 64);         // prefetch next block
    if (more) wait_async_le4(); else wait_async_le0();
    __syncthreads();                           // cur tiles visible to all waves

    const _Float16* kt = (const _Float16*)(smem + KOFF(cur));
    const _Float16* vt = (const _Float16*)(smem + VOFF(cur));

    // ---- S = (Q K^T) * scale, causal-masked: 4 tiles of 16 key cols ----
    v8f s[4];
#pragma unroll
    for (int ni = 0; ni < 4; ++ni) {
      const _Float16* kp = kt + ni * 16 * 80;
      v16h b0 = load_frag_b(kp,      80);      // d 0..31
      v16h b1 = load_frag_b(kp + 32, 80);      // d 32..63
      v8f z = {};
      z = wmma_step(aq0, b0, z);
      z = wmma_step(aq1, b1, z);
      s[ni] = z;
    }

    // ---- online softmax update (per row r, cross-lane within half) ----
#pragma unroll
    for (int r = 0; r < 8; ++r) {
      int row = qbase + r + ro;
      float mx = -3.0e38f;
#pragma unroll
      for (int ni = 0; ni < 4; ++ni) {
        int col = kb + ni * 16 + lc;
        float v = s[ni][r] * 0.125f;           // 1/sqrt(64)
        v = (col > row) ? -3.0e38f : v;
        s[ni][r] = v;
        mx = fmaxf(mx, v);
      }
#pragma unroll
      for (int off = 1; off < 16; off <<= 1)
        mx = fmaxf(mx, __shfl_xor(mx, off, 32));
      float mnew  = fmaxf(mrow[r], mx);
      float alpha = __expf(mrow[r] - mnew);
      float psum  = 0.0f;
#pragma unroll
      for (int ni = 0; ni < 4; ++ni) {
        float p = __expf(s[ni][r] - mnew);
        s[ni][r] = p;
        psum += p;
      }
#pragma unroll
      for (int off = 1; off < 16; off <<= 1)
        psum += __shfl_xor(psum, off, 32);
      lrow[r] = lrow[r] * alpha + psum;
      mrow[r] = mnew;
#pragma unroll
      for (int jd = 0; jd < 4; ++jd) o[jd][r] *= alpha;
    }

    // ---- P: C-layout -> LDS -> A-layout (wave-private region) ----
#pragma unroll
    for (int ni = 0; ni < 4; ++ni)
#pragma unroll
      for (int r = 0; r < 8; ++r)
        pwv[(r + ro) * 80 + ni * 16 + lc] = (_Float16)s[ni][r];

    v16h ap0 = load_frag_a(pwv,      80);      // keys kb..kb+31
    v16h ap1 = load_frag_a(pwv + 32, 80);      // keys kb+32..kb+63

    // ---- O += P V : B-fragments key-contiguous from LDS V tile ----
#pragma unroll
    for (int jd = 0; jd < 4; ++jd) {
      const _Float16* vp = vt + jd * 16 * 80;
      v16h bv0 = load_frag_b(vp,      80);     // keys 0..31 of block
      v16h bv1 = load_frag_b(vp + 32, 80);     // keys 32..63 of block
      o[jd] = wmma_step(ap0, bv0, o[jd]);
      o[jd] = wmma_step(ap1, bv1, o[jd]);
    }
    __syncthreads();   // all waves done with cur before it is restaged
  }

  // ---- normalize, write merged [B,T,C] layout for the out-projection ----
#pragma unroll
  for (int jd = 0; jd < 4; ++jd)
#pragma unroll
    for (int r = 0; r < 8; ++r) {
      int row = qbase + r + ro;
      float val = o[jd][r] / lrow[r];
      Y[((size_t)b * T_SEQ + row) * C_DIM + h * D_HEAD + jd * 16 + lc] =
          (_Float16)val;
    }
}

// ---------------------------------------------------------------------------
// Host-side orchestration
// ---------------------------------------------------------------------------
extern "C" void kernel_launch(void* const* d_in, const int* in_sizes, int n_in,
                              void* d_out, int out_size, void* d_ws, size_t ws_size,
                              hipStream_t stream) {
  (void)in_sizes; (void)n_in; (void)out_size; (void)ws_size;

  const float* x  = (const float*)d_in[0];
  const float* wq = (const float*)d_in[1];
  const float* bq = (const float*)d_in[2];
  const float* wk = (const float*)d_in[3];
  const float* bk = (const float*)d_in[4];
  const float* wv = (const float*)d_in[5];
  const float* bv = (const float*)d_in[6];
  const float* wp = (const float*)d_in[7];
  const float* bp = (const float*)d_in[8];
  float* out = (float*)d_out;

  const long long MC = (long long)M_ROWS * C_DIM;      // 8,388,608
  const long long WW = (long long)C_DIM * C_DIM;       // 1,048,576

  char* ws = (char*)d_ws;
  _Float16* xh  = (_Float16*)(ws);                     // 16 MB
  _Float16* wqh = (_Float16*)(ws + 16777216);          //  2 MB
  _Float16* wkh = (_Float16*)(ws + 18874368);
  _Float16* wvh = (_Float16*)(ws + 20971520);
  _Float16* wph = (_Float16*)(ws + 23068672);
  _Float16* qf  = (_Float16*)(ws + 25165824);          // 16 MB
  _Float16* kf  = (_Float16*)(ws + 41943040);          // 16 MB
  _Float16* vf  = (_Float16*)(ws + 58720256);          // 16 MB
  _Float16* Kr  = (_Float16*)(ws + 75497472);          // 16 MB
  _Float16* Vt  = (_Float16*)(ws + 92274688);          // 16 MB
  _Float16* Qr  = xh;   // x-half dead after the 3 QKV GEMMs
  _Float16* yh  = qf;   // q-proj buffer dead after RoPE

  // 1) downcast inputs to f16 (L2-resident afterwards)
  cvt_f16_kernel<<<4096, 256, 0, stream>>>(x,  xh,  MC);
  cvt_f16_kernel<<<1024, 256, 0, stream>>>(wq, wqh, WW);
  cvt_f16_kernel<<<1024, 256, 0, stream>>>(wk, wkh, WW);
  cvt_f16_kernel<<<1024, 256, 0, stream>>>(wv, wvh, WW);
  cvt_f16_kernel<<<1024, 256, 0, stream>>>(wp, wph, WW);

  // 2) Q/K/V projections (WMMA, fp32 accumulate, f16 out)
  dim3 ggrid(C_DIM / 128, M_ROWS / 128);               // (8, 64)
  gemm_bias_kernel<<<ggrid, 256, 0, stream>>>(xh, wqh, bq, qf, nullptr,
                                              M_ROWS, C_DIM, C_DIM, 0);
  gemm_bias_kernel<<<ggrid, 256, 0, stream>>>(xh, wkh, bk, kf, nullptr,
                                              M_ROWS, C_DIM, C_DIM, 0);
  gemm_bias_kernel<<<ggrid, 256, 0, stream>>>(xh, wvh, bv, vf, nullptr,
                                              M_ROWS, C_DIM, C_DIM, 0);

  // 3) RoPE + head split / V transpose
  rope_split_kernel<<<16384, 256, 0, stream>>>(qf, kf, vf, Qr, Kr, Vt);

  // 4) causal flash attention (double-buffered async K/V staging, 60 KB LDS)
  dim3 agrid(T_SEQ / 128, 4 * N_HEAD);                 // (16, 64)
  flash_attn_kernel<<<agrid, 256, SMEM_BYTES, stream>>>(Qr, Kr, Vt, yh);

  // 5) output projection (fp32 epilogue straight into d_out)
  gemm_bias_kernel<<<ggrid, 256, 0, stream>>>(yh, wph, bp, nullptr, out,
                                              M_ROWS, C_DIM, C_DIM, 1);
}
